// FP8SwiGLUMLP_16638703304736
// MI455X (gfx1250) — 8-bit matrix operand order repaired and checked by reading, compiled, not run on hardware
//
#include <hip/hip_runtime.h>
#include <cstdint>
#include <cstddef>


typedef __attribute__((ext_vector_type(16))) int   v16i;
typedef __attribute__((ext_vector_type(8)))  float v8f;

#define H_DIM 4096

#define S_WAIT_ASYNC_0() asm volatile("s_wait_asynccnt 0x0" ::: "memory")
#define S_WAIT_ASYNC_2() asm volatile("s_wait_asynccnt 0x2" ::: "memory")
#define S_WAIT_ASYNC_4() asm volatile("s_wait_asynccnt 0x4" ::: "memory")

__device__ __forceinline__ unsigned int f32_to_e4m3(float f) {
    unsigned int u    = __float_as_uint(f);
    unsigned int sign = (u >> 24) & 0x80u;
    unsigned int au   = u & 0x7FFFFFFFu;
    if (au > 0x43E00000u) au = 0x43E00000u;
    if (au < 0x3C800000u) {
        float a = __uint_as_float(au);
        unsigned int d = (unsigned int)__float2int_rn(a * 512.0f);
        return sign | d;
    }
    unsigned int r = au + 0x0007FFFFu + ((au >> 20) & 1u);
    unsigned int e = (r >> 23) - 120u;
    unsigned int m = (r >> 20) & 7u;
    unsigned int code = (e << 3) | m;
    if (code > 0x7Eu) code = 0x7Eu;
    return sign | code;
}

__global__ __launch_bounds__(256)
void quant_fp8_kernel(const float* __restrict__ src, unsigned char* __restrict__ dst,
                      const float* __restrict__ scale_ptr, size_t n4) {
    const float inv = scale_ptr ? (1.0f / scale_ptr[0]) : 1.0f;
    size_t i = (size_t)blockIdx.x * blockDim.x + threadIdx.x;
    const size_t stride = (size_t)gridDim.x * blockDim.x;
    const float4* s4 = (const float4*)src;
    unsigned int* d4 = (unsigned int*)dst;
    for (; i < n4; i += stride) {
        float4 v = s4[i];
        unsigned int b0 = f32_to_e4m3(v.x * inv);
        unsigned int b1 = f32_to_e4m3(v.y * inv);
        unsigned int b2 = f32_to_e4m3(v.z * inv);
        unsigned int b3 = f32_to_e4m3(v.w * inv);
        d4[i] = b0 | (b1 << 8) | (b2 << 16) | (b3 << 24);
    }
}

__device__ __forceinline__ void stage_tile_async(const unsigned char* __restrict__ G,
                                                 int ld, int row0, int k0,
                                                 unsigned char* sdst) {
    const int t = threadIdx.x;
#pragma unroll
    for (int c = 0; c < 2; ++c) {
        const int chunk = t + c * 256;
        const int row   = chunk >> 3;
        const int kc    = (chunk & 7) << 4;
        const unsigned char* g = G + (size_t)(row0 + row) * (size_t)ld + (k0 + kc);
        const unsigned lofs = (unsigned)(uintptr_t)(sdst + row * 128 + kc);
        asm volatile("global_load_async_to_lds_b128 %0, %1, off"
                     :: "v"(lofs), "v"(g) : "memory");
    }
}

__device__ __forceinline__ v16i lds_a_frag(const unsigned char* tile, int m0, int lane) {
    const int row = m0 + (lane & 15);
    const int kb  = (lane >> 4) << 4;
    const unsigned char* p = tile + row * 128 + kb;
    v16i a;
#pragma unroll
    for (int i = 0; i < 8; ++i) {
        int2 d = *(const int2*)(p + 32 * (i >> 1) + 8 * (i & 1));
        a[2 * i]     = d.x;
        a[2 * i + 1] = d.y;
    }
    return a;
}

__device__ __forceinline__ v16i load_b_frag(const unsigned char* __restrict__ Bt,
                                            int ldb, int n0, int k0, int lane) {
    const int col = n0 + (lane & 15);
    const int kb  = k0 + ((lane >> 4) << 4);
    const unsigned char* p = Bt + (size_t)col * (size_t)ldb + kb;
    v16i b;
#pragma unroll
    for (int g = 0; g < 4; ++g) {
        int4 d = *(const int4*)(p + 32 * g);
        b[4 * g]     = d.x;
        b[4 * g + 1] = d.y;
        b[4 * g + 2] = d.z;
        b[4 * g + 3] = d.w;
    }
    return b;
}

__device__ __forceinline__ v8f wmma_fp8(v16i a, v16i b, v8f c) {
    return __builtin_amdgcn_wmma_f32_16x16x128_fp8_fp8(a, b, (short)0, c, false, false);
}

__global__ __launch_bounds__(256)
void gemm1_swiglu_fp8(const unsigned char* __restrict__ Xg,
                      const unsigned char* __restrict__ Xu,
                      const unsigned char* __restrict__ Gq,
                      const unsigned char* __restrict__ Uq,
                      unsigned char* __restrict__ Hq,
                      const float* __restrict__ gw_s, const float* __restrict__ uw_s,
                      const float* __restrict__ gin_s, const float* __restrict__ uin_s,
                      const float* __restrict__ din_s,
                      int M, int Hdim, int Idim) {
    __shared__ unsigned char sAg[2][64 * 128];
    __shared__ unsigned char sAu[2][64 * 128];

    const int lane = threadIdx.x & 31;
    const int wave = threadIdx.x >> 5;
    const int mloc = (wave >> 2) * 32;
    const int mblk = blockIdx.y * 64;
    const int n0   = blockIdx.x * 128 + (wave & 3) * 32;

    v8f cg[2][2], cu[2][2];
#pragma unroll
    for (int i = 0; i < 2; ++i)
#pragma unroll
        for (int j = 0; j < 2; ++j) { cg[i][j] = (v8f){}; cu[i][j] = (v8f){}; }

    const int nk = Hdim / 128;
    stage_tile_async(Xg, Hdim, mblk, 0, &sAg[0][0]);
    stage_tile_async(Xu, Hdim, mblk, 0, &sAu[0][0]);

    for (int ks = 0; ks < nk; ++ks) {
        const int buf = ks & 1;
        const int k0  = ks * 128;
        if (ks + 1 < nk) {
            stage_tile_async(Xg, Hdim, mblk, k0 + 128, &sAg[buf ^ 1][0]);
            stage_tile_async(Xu, Hdim, mblk, k0 + 128, &sAu[buf ^ 1][0]);
            __builtin_prefetch(Gq + (size_t)(n0 + (lane & 15)) * Hdim + k0 + 128, 0, 1);
            __builtin_prefetch(Uq + (size_t)(n0 + (lane & 15)) * Hdim + k0 + 128, 0, 1);
            S_WAIT_ASYNC_4();
        } else {
            S_WAIT_ASYNC_0();
        }
        __syncthreads();

        v16i a0g = lds_a_frag(&sAg[buf][0], mloc,      lane);
        v16i a1g = lds_a_frag(&sAg[buf][0], mloc + 16, lane);
        v16i a0u = lds_a_frag(&sAu[buf][0], mloc,      lane);
        v16i a1u = lds_a_frag(&sAu[buf][0], mloc + 16, lane);
#pragma unroll
        for (int nt = 0; nt < 2; ++nt) {
            v16i bg = load_b_frag(Gq, Hdim, n0 + nt * 16, k0, lane);
            v16i bu = load_b_frag(Uq, Hdim, n0 + nt * 16, k0, lane);
            cg[0][nt] = wmma_fp8(a0g, bg, cg[0][nt]);
            cg[1][nt] = wmma_fp8(a1g, bg, cg[1][nt]);
            cu[0][nt] = wmma_fp8(a0u, bu, cu[0][nt]);
            cu[1][nt] = wmma_fp8(a1u, bu, cu[1][nt]);
        }
        __syncthreads();
    }

    const float gsc   = gw_s[0] * gin_s[0];
    const float usc   = uw_s[0] * uin_s[0];
    const float inv_d = 1.0f / din_s[0];

#pragma unroll
    for (int mt = 0; mt < 2; ++mt)
#pragma unroll
        for (int nt = 0; nt < 2; ++nt)
#pragma unroll
            for (int j = 0; j < 8; ++j) {
                const int row = mblk + mloc + mt * 16 + j + ((lane >> 4) << 3);
                const int col = n0 + nt * 16 + (lane & 15);
                const float g = cg[mt][nt][j] * gsc;
                const float u = cu[mt][nt][j] * usc;
                const float h = (g / (1.0f + __expf(-g))) * u;
                Hq[(size_t)row * (size_t)Idim + col] = (unsigned char)f32_to_e4m3(h * inv_d);
            }
}

__global__ __launch_bounds__(256)
void gemm2_fp8(const unsigned char* __restrict__ Hq,
               const unsigned char* __restrict__ Dq,
               float* __restrict__ Out,
               const float* __restrict__ dw_s, const float* __restrict__ din_s,
               int M, int Idim, int Hdim) {
    __shared__ unsigned char sA[2][64 * 128];

    const int lane = threadIdx.x & 31;
    const int wave = threadIdx.x >> 5;
    const int mloc = (wave >> 2) * 32;
    const int mblk = blockIdx.y * 64;
    const int n0   = blockIdx.x * 128 + (wave & 3) * 32;

    v8f c[2][2];
#pragma unroll
    for (int i = 0; i < 2; ++i)
#pragma unroll
        for (int j = 0; j < 2; ++j) c[i][j] = (v8f){};

    const int nk = Idim / 128;
    stage_tile_async(Hq, Idim, mblk, 0, &sA[0][0]);

    for (int ks = 0; ks < nk; ++ks) {
        const int buf = ks & 1;
        const int k0  = ks * 128;
        if (ks + 1 < nk) {
            stage_tile_async(Hq, Idim, mblk, k0 + 128, &sA[buf ^ 1][0]);
            __builtin_prefetch(Dq + (size_t)(n0 + (lane & 15)) * Idim + k0 + 128, 0, 1);
            S_WAIT_ASYNC_2();
        } else {
            S_WAIT_ASYNC_0();
        }
        __syncthreads();

        v16i a0 = lds_a_frag(&sA[buf][0], mloc,      lane);
        v16i a1 = lds_a_frag(&sA[buf][0], mloc + 16, lane);
#pragma unroll
        for (int nt = 0; nt < 2; ++nt) {
            v16i b = load_b_frag(Dq, Idim, n0 + nt * 16, k0, lane);
            c[0][nt] = wmma_fp8(a0, b, c[0][nt]);
            c[1][nt] = wmma_fp8(a1, b, c[1][nt]);
        }
        __syncthreads();
    }

    const float sc = dw_s[0] * din_s[0];
#pragma unroll
    for (int mt = 0; mt < 2; ++mt)
#pragma unroll
        for (int nt = 0; nt < 2; ++nt)
#pragma unroll
            for (int j = 0; j < 8; ++j) {
                const int row = mblk + mloc + mt * 16 + j + ((lane >> 4) << 3);
                const int col = n0 + nt * 16 + (lane & 15);
                Out[(size_t)row * (size_t)Hdim + col] = c[mt][nt][j] * sc;
            }
}

extern "C" void kernel_launch(void* const* d_in, const int* in_sizes, int n_in,
                              void* d_out, int out_size, void* d_ws, size_t ws_size,
                              hipStream_t stream) {
    const float* x      = (const float*)d_in[0];
    const float* gate_w = (const float*)d_in[1];
    const float* up_w   = (const float*)d_in[2];
    const float* down_w = (const float*)d_in[3];
    const float* gw_s   = (const float*)d_in[4];
    const float* uw_s   = (const float*)d_in[5];
    const float* dw_s   = (const float*)d_in[6];
    const float* gin_s  = (const float*)d_in[7];
    const float* uin_s  = (const float*)d_in[8];
    const float* din_s  = (const float*)d_in[9];

    const int Hd = H_DIM;
    const int Id = (int)((size_t)in_sizes[1] / Hd);
    const int M  = (int)((size_t)in_sizes[0] / Hd);

    unsigned char* ws  = (unsigned char*)d_ws;
    unsigned char* xgq = ws;
    unsigned char* xuq = xgq + (size_t)M * Hd;
    unsigned char* gq  = xuq + (size_t)M * Hd;
    unsigned char* uq  = gq  + (size_t)Id * Hd;
    unsigned char* dq  = uq  + (size_t)Id * Hd;
    unsigned char* hq  = dq  + (size_t)Hd * Id;

    {
        size_t n4;
        n4 = ((size_t)M * Hd) / 4;
        quant_fp8_kernel<<<(unsigned)((n4 + 255) / 256), 256, 0, stream>>>(x, xgq, gin_s, n4);
        quant_fp8_kernel<<<(unsigned)((n4 + 255) / 256), 256, 0, stream>>>(x, xuq, uin_s, n4);
        n4 = ((size_t)Id * Hd) / 4;
        quant_fp8_kernel<<<(unsigned)((n4 + 255) / 256), 256, 0, stream>>>(gate_w, gq, nullptr, n4);
        quant_fp8_kernel<<<(unsigned)((n4 + 255) / 256), 256, 0, stream>>>(up_w,   uq, nullptr, n4);
        quant_fp8_kernel<<<(unsigned)((n4 + 255) / 256), 256, 0, stream>>>(down_w, dq, nullptr, n4);
    }

    {
        dim3 grid(Id / 128, M / 64);
        gemm1_swiglu_fp8<<<grid, 256, 0, stream>>>(xgq, xuq, gq, uq, hq,
                                                   gw_s, uw_s, gin_s, uin_s, din_s,
                                                   M, Hd, Id);
    }

    {
        dim3 grid(Hd / 128, M / 64);
        gemm2_fp8<<<grid, 256, 0, stream>>>(hq, dq, (float*)d_out, dw_s, din_s, M, Id, Hd);
    }
}
